// LSTM_43739946943027
// MI455X (gfx1250) — matrix operand order repaired and checked by reading, not compiled, not run on hardware
//
#include <hip/hip_runtime.h>
#include <hip/hip_bf16.h>

// ---------------- problem constants ----------------
#define T_STEPS   1024
#define BATCH     64
#define H_REAL    500
#define HPAD      512            // padded hidden size (K half / N-per-gate)
#define KCOMB     1024           // combined K = [input(512) ; recurrent(512)]
#define NWG       16             // workgroups (== max cluster size)
#define NT_PER_WG 8              // 4 gates x 2 hidden-tiles of 16 cols
#define WSWZ_WG   (NT_PER_WG * 32 * 32 * 16)     // bf16 elems per WG slice
#define WSWZ_LAYER ((size_t)NWG * WSWZ_WG)       // 2,097,152 bf16 = 4 MB

typedef __attribute__((ext_vector_type(16))) __bf16    v16bf;
typedef __attribute__((ext_vector_type(8)))  float     v8f;
typedef __attribute__((ext_vector_type(4)))  unsigned  u32x4;

union ABv { v16bf v; u32x4 q[2]; };

static __device__ inline v8f vzero8() {
  v8f v;
#pragma unroll
  for (int i = 0; i < 8; ++i) v[i] = 0.f;
  return v;
}
static __device__ inline float fast_sig(float x) { return 1.f / (1.f + __expf(-x)); }

// ---------------- prep: weights into WMMA-B operand order ----------------
// Layout: [layer][wg][nt][kstep][lane][16 elems], nt = gate*2 + htile.
// B operand (32x16 bf16): lanes 0-15 hold K={0..7,16..23}, lanes 16-31 hold K={8..15,24..31},
// element e within a lane = K (e<8 ? klo+e : 16+klo+e-8), klo = (lane<16 ? 0 : 8), N = lane&15.
__global__ void prep_wswz(__bf16* __restrict__ wswz,
                          const float* __restrict__ wih0, const float* __restrict__ whh0,
                          const float* __restrict__ wih1, const float* __restrict__ whh1,
                          const float* __restrict__ wih2, const float* __restrict__ whh2) {
  size_t id = (size_t)blockIdx.x * 256 + threadIdx.x;
  if (id >= 3 * WSWZ_LAYER) return;
  int layer = (int)(id / WSWZ_LAYER);
  int r     = (int)(id % WSWZ_LAYER);
  int e    = r & 15;
  int lane = (r >> 4) & 31;
  int ks   = (r >> 9) & 31;
  int nt   = (r >> 14) & 7;
  int wg   = (r >> 17) & 15;

  const float* wih; const float* whh; int din;
  if (layer == 0)      { wih = wih0; whh = whh0; din = 1;      }
  else if (layer == 1) { wih = wih1; whh = whh1; din = H_REAL; }
  else                 { wih = wih2; whh = whh2; din = H_REAL; }

  int gate = nt >> 1;
  int ht   = nt & 1;
  int ncol = wg * 32 + ht * 16 + (lane & 15);          // hidden col 0..511
  int k    = ks * 32 + ((lane < 16) ? 0 : 8) + ((e < 8) ? e : e + 8);     // combined K 0..1023

  float val = 0.f;
  if (ncol < H_REAL) {
    int g = gate * H_REAL + ncol;                      // gate row in 4H
    if (k < HPAD) { if (k < din) val = wih[(size_t)g * din + k]; }
    else { int kr = k - HPAD; if (kr < H_REAL) val = whh[(size_t)g * H_REAL + kr]; }
  }
  wswz[id] = (__bf16)val;
}

// ---------------- prep: layer-0 input stream [T][B][HPAD] bf16 ----------------
__global__ void prep_in0(__bf16* __restrict__ in0, const float* __restrict__ x) {
  size_t id = (size_t)blockIdx.x * 256 + threadIdx.x;
  if (id >= (size_t)T_STEPS * BATCH * HPAD) return;
  int k = (int)(id & (HPAD - 1));
  int b = (int)((id >> 9) & (BATCH - 1));
  int t = (int)(id >> 15);
  in0[id] = (__bf16)((k == 0) ? x[(size_t)b * T_STEPS + t] : 0.f);
}

__global__ void zero_bf16(__bf16* __restrict__ p, int n) {
  for (int i = blockIdx.x * 256 + threadIdx.x; i < n; i += gridDim.x * 256)
    p[i] = (__bf16)0.f;
}
__global__ void zero_u32(unsigned* __restrict__ p, int n) {
  int i = threadIdx.x;
  if (i < n) p[i] = 0u;
}
// bias[4][512] = b_ih + b_hh per gate, zero-padded cols
__global__ void make_bias(float* __restrict__ dst,
                          const float* __restrict__ bih, const float* __restrict__ bhh) {
  int id = blockIdx.x * 256 + threadIdx.x;
  if (id >= 4 * HPAD) return;
  int gate = id >> 9, col = id & (HPAD - 1);
  dst[id] = (col < H_REAL) ? (bih[gate * H_REAL + col] + bhh[gate * H_REAL + col]) : 0.f;
}

// ---------------- per-timestep grid barrier ----------------
__device__ inline void step_barrier(unsigned* cnt) {
  __threadfence();
  __syncthreads();
#if __has_builtin(__builtin_amdgcn_cluster_id_x) && __has_builtin(__builtin_amdgcn_s_cluster_barrier)
  if (__builtin_amdgcn_cluster_id_x() != 0) {     // launched as a real cluster: HW barrier
    __builtin_amdgcn_s_cluster_barrier();
    __syncthreads();
    __threadfence();
    return;
  }
#endif
  if (threadIdx.x == 0) {                         // fallback: atomic phase barrier
    unsigned ticket = atomicAdd(cnt, 1u);
    unsigned target = (ticket / NWG + 1u) * NWG;
    while (atomicAdd(cnt, 0u) < target) __builtin_amdgcn_s_sleep(2);
  }
  __syncthreads();
  __threadfence();
}

// ---------------- fused LSTM layer (recurrent scan) ----------------
// in_seq : [T][B][HPAD] bf16  (layer input, h of layer below or x-projection)
// out_seq: [T+1][B][HPAD] bf16, slot 0 pre-zeroed; h_t written to slot t+1
// Grid: 16 WGs x 256 thr. WG wg owns hidden cols [wg*32, wg*32+32).
// Wave (mt,ht): batch tile mt (16 rows), hidden sub-tile ht (16 cols), all 4 gates.
__global__ __launch_bounds__(256, 1) void lstm_layer_kernel(
    const __bf16* __restrict__ in_seq, __bf16* __restrict__ out_seq,
    const __bf16* __restrict__ wsl, const float* __restrict__ bias,
    unsigned* __restrict__ sync_cnt) {
  extern __shared__ __bf16 lds_w[];                // 8 nt x 32 ks x 32 lanes x 16 = 256 KB
  const int tid = threadIdx.x;
  const int wg  = blockIdx.x;

  { // stage this WG's weight slice into LDS once; reused for all 1024 steps
    const u32x4* src = (const u32x4*)(wsl + (size_t)wg * WSWZ_WG);
    u32x4* dst = (u32x4*)lds_w;
#pragma unroll 4
    for (int i = tid; i < WSWZ_WG / 8; i += 256) dst[i] = src[i];
  }
  __syncthreads();

  const int lane = tid & 31;
  const int wave = tid >> 5;
  const int mt   = wave >> 1;                      // 0..3 batch tile
  const int ht   = wave & 1;                       // 0..1 hidden sub-tile
  const int row  = mt * 16 + (lane & 15);          // batch row (A-matrix M)
  const int c0   = (lane & 16) ? 8 : 0;            // A 16-bit layout K sub-offset
  const int hcol = wg * 32 + ht * 16 + (lane & 15);
  const int rbase = mt * 16 + ((lane & 16) ? 8 : 0);

  const float bI = bias[0 * HPAD + hcol], bF = bias[1 * HPAD + hcol],
              bG = bias[2 * HPAD + hcol], bO = bias[3 * HPAD + hcol];

  v8f cst = vzero8();                              // cell-state tile lives in VGPRs
  for (int t = 0; t < T_STEPS; ++t) {
    const __bf16* aLo = in_seq  + ((size_t)t * BATCH + row) * HPAD;  // K 0..511
    const __bf16* aHi = out_seq + ((size_t)t * BATCH + row) * HPAD;  // K 512..1023 (h_{t-1})
    v8f accI = vzero8(), accF = vzero8(), accG = vzero8(), accO = vzero8();

#pragma unroll 4
    for (int ks = 0; ks < 32; ++ks) {
      const __bf16* ab = (ks < 16) ? (aLo + ks * 32) : (aHi + (ks - 16) * 32);
      ABv a;
      a.q[0] = *(const u32x4*)(ab + c0);           // K runs {c0..c0+7, c0+16..c0+23}
      a.q[1] = *(const u32x4*)(ab + c0 + 16);
      ABv b;
      { const u32x4* bp = (const u32x4*)&lds_w[(((0 * 2 + ht) * 32 + ks) * 32 + lane) * 16];
        b.q[0] = bp[0]; b.q[1] = bp[1];
        accI = __builtin_amdgcn_wmma_f32_16x16x32_bf16(false, a.v, false, b.v, (short)0, accI, false, false); }
      { const u32x4* bp = (const u32x4*)&lds_w[(((1 * 2 + ht) * 32 + ks) * 32 + lane) * 16];
        b.q[0] = bp[0]; b.q[1] = bp[1];
        accF = __builtin_amdgcn_wmma_f32_16x16x32_bf16(false, a.v, false, b.v, (short)0, accF, false, false); }
      { const u32x4* bp = (const u32x4*)&lds_w[(((2 * 2 + ht) * 32 + ks) * 32 + lane) * 16];
        b.q[0] = bp[0]; b.q[1] = bp[1];
        accG = __builtin_amdgcn_wmma_f32_16x16x32_bf16(false, a.v, false, b.v, (short)0, accG, false, false); }
      { const u32x4* bp = (const u32x4*)&lds_w[(((3 * 2 + ht) * 32 + ks) * 32 + lane) * 16];
        b.q[0] = bp[0]; b.q[1] = bp[1];
        accO = __builtin_amdgcn_wmma_f32_16x16x32_bf16(false, a.v, false, b.v, (short)0, accO, false, false); }
    }

    // gate nonlinearity + cell update, fp32 in registers; write h_t (bf16)
    __bf16* outp = out_seq + (size_t)(t + 1) * BATCH * HPAD;
#pragma unroll
    for (int e = 0; e < 8; ++e) {
      float iv = fast_sig(accI[e] + bI);
      float fv = fast_sig(accF[e] + bF);
      float gv = tanhf(accG[e] + bG);
      float ov = fast_sig(accO[e] + bO);
      float c  = fv * cst[e] + iv * gv;
      cst[e]   = c;
      float h  = ov * tanhf(c);
      outp[(size_t)(rbase + e) * HPAD + hcol] = (__bf16)h;
    }
    if (t + 1 < T_STEPS)
      __builtin_prefetch((const void*)(in_seq + ((size_t)(t + 1) * BATCH + row) * HPAD), 0, 1);
    step_barrier(sync_cnt);                        // h_t visible before any WG starts t+1
  }
}

// ---------------- final linear on last timestep ----------------
__global__ void final_kernel(const __bf16* __restrict__ hbuf,
                             const float* __restrict__ lin_w, const float* __restrict__ lin_b,
                             float* __restrict__ out) {
  __shared__ float red[8];
  int b = blockIdx.x, tid = threadIdx.x;
  const __bf16* h = hbuf + ((size_t)T_STEPS * BATCH + b) * HPAD;  // slot T = h_{T-1}
  float s = 0.f;
  for (int k = tid; k < H_REAL; k += 256) s += (float)h[k] * lin_w[k];
#pragma unroll
  for (int o = 16; o > 0; o >>= 1) s += __shfl_down(s, o, 32);
  if ((tid & 31) == 0) red[tid >> 5] = s;
  __syncthreads();
  if (tid == 0) {
    float tot = lin_b[0];
    for (int i = 0; i < 8; ++i) tot += red[i];
    out[b] = tot;
  }
}

// ---------------- host orchestration ----------------
extern "C" void kernel_launch(void* const* d_in, const int* in_sizes, int n_in,
                              void* d_out, int out_size, void* d_ws, size_t ws_size,
                              hipStream_t stream) {
  (void)in_sizes; (void)n_in; (void)out_size; (void)ws_size;
  const float* x     = (const float*)d_in[0];
  const float* wih0  = (const float*)d_in[1];
  const float* whh0  = (const float*)d_in[2];
  const float* bih0  = (const float*)d_in[3];
  const float* bhh0  = (const float*)d_in[4];
  const float* wih1  = (const float*)d_in[5];
  const float* whh1  = (const float*)d_in[6];
  const float* bih1  = (const float*)d_in[7];
  const float* bhh1  = (const float*)d_in[8];
  const float* wih2  = (const float*)d_in[9];
  const float* whh2  = (const float*)d_in[10];
  const float* bih2  = (const float*)d_in[11];
  const float* bhh2  = (const float*)d_in[12];
  const float* lin_w = (const float*)d_in[13];
  const float* lin_b = (const float*)d_in[14];

  char* base = (char*)d_ws;
  size_t off = 0;
  auto take = [&](size_t bytes) { size_t o = off; off = (off + bytes + 255) & ~(size_t)255; return o; };
  __bf16*   wswz = (__bf16*)(base + take(3 * WSWZ_LAYER * sizeof(__bf16)));
  float*    bias = (float*)(base + take((size_t)3 * 4 * HPAD * sizeof(float)));
  __bf16*   in0  = (__bf16*)(base + take((size_t)T_STEPS * BATCH * HPAD * sizeof(__bf16)));
  __bf16*   bufA = (__bf16*)(base + take((size_t)(T_STEPS + 1) * BATCH * HPAD * sizeof(__bf16)));
  __bf16*   bufB = (__bf16*)(base + take((size_t)(T_STEPS + 1) * BATCH * HPAD * sizeof(__bf16)));
  unsigned* sync = (unsigned*)(base + take(256));

  // prep (rebuilt every call: deterministic, graph-capture safe)
  prep_wswz<<<(unsigned)((3 * WSWZ_LAYER + 255) / 256), 256, 0, stream>>>(
      wswz, wih0, whh0, wih1, whh1, wih2, whh2);
  prep_in0<<<(unsigned)(((size_t)T_STEPS * BATCH * HPAD + 255) / 256), 256, 0, stream>>>(in0, x);
  zero_bf16<<<128, 256, 0, stream>>>(bufA, BATCH * HPAD);   // slot 0 = h_{-1} = 0
  zero_bf16<<<128, 256, 0, stream>>>(bufB, BATCH * HPAD);
  make_bias<<<8, 256, 0, stream>>>(bias + 0 * 4 * HPAD, bih0, bhh0);
  make_bias<<<8, 256, 0, stream>>>(bias + 1 * 4 * HPAD, bih1, bhh1);
  make_bias<<<8, 256, 0, stream>>>(bias + 2 * 4 * HPAD, bih2, bhh2);
  zero_u32<<<1, 32, 0, stream>>>(sync, 8);

  const size_t LDSB = (size_t)WSWZ_WG * sizeof(__bf16);     // 256 KB dynamic LDS
  lstm_layer_kernel<<<NWG, 256, LDSB, stream>>>(in0, bufA, wswz, bias + 0 * 4 * HPAD, sync + 0);
  lstm_layer_kernel<<<NWG, 256, LDSB, stream>>>(bufA + (size_t)BATCH * HPAD, bufB,
                                                wswz + WSWZ_LAYER, bias + 1 * 4 * HPAD, sync + 1);
  lstm_layer_kernel<<<NWG, 256, LDSB, stream>>>(bufB + (size_t)BATCH * HPAD, bufA,
                                                wswz + 2 * WSWZ_LAYER, bias + 2 * 4 * HPAD, sync + 2);
  final_kernel<<<BATCH, 256, 0, stream>>>(bufA, lin_w, lin_b, (float*)d_out);
}
